// DiT_89970974916630
// MI455X (gfx1250) — matrix operand order repaired and checked by reading, not compiled, not run on hardware
//
#include <hip/hip_runtime.h>
#include <math.h>

typedef __bf16 bf16;
typedef __attribute__((ext_vector_type(16))) __bf16 v16bf;
typedef __attribute__((ext_vector_type(8)))  float  v8f;

__device__ inline v8f zero_v8f() {
    v8f z = {0.f, 0.f, 0.f, 0.f, 0.f, 0.f, 0.f, 0.f};
    return z;
}

// Wave-relative LDS byte offset of a __shared__ pointer: the generic (flat)
// address of LDS is {aperture_hi, 32'lds_offset}; truncation gives the offset
// the async-to-LDS VDST VGPR expects.
__device__ inline uint32_t lds_off_of(const void* p) {
    return (uint32_t)(uintptr_t)p;
}

// gfx1250 async global->LDS copy, 16B per lane, tracked by ASYNCcnt.
__device__ inline void async_copy_b128(uint32_t lds_off, const void* gaddr) {
    asm volatile("global_load_async_to_lds_b128 %0, %1, off"
                 :: "v"(lds_off), "v"(gaddr)
                 : "memory");
}

// ---------------------------------------------------------------------------
// Patchify + interleaved sincos positional embedding.
// ---------------------------------------------------------------------------
__global__ __launch_bounds__(256) void patchify_kernel(const float* __restrict__ x,
                                                       const float* __restrict__ cw,
                                                       const float* __restrict__ cb,
                                                       float* __restrict__ h) {
    const int row = blockIdx.x;              // b*1024 + tok
    const int b   = row >> 10;
    const int tok = row & 1023;
    const int gh  = tok >> 5, gw = tok & 31;
    __shared__ float patch[192];             // [c][p][q] matches conv_w inner layout
    for (int i = threadIdx.x; i < 192; i += 256) {
        int c = i >> 6, rem = i & 63, p = rem >> 3, q = rem & 7;
        patch[i] = x[(((size_t)b * 3 + c) * 256 + gh * 8 + p) * 256 + gw * 8 + q];
    }
    __syncthreads();
    for (int d = threadIdx.x; d < 768; d += 256) {
        float s = cb[d];
        const float* w = cw + (size_t)d * 192;
        #pragma unroll 4
        for (int i = 0; i < 192; ++i) s += patch[i] * w[i];
        int i2 = d >> 1;
        float fr = __powf(10000.f, -(2.f * (float)i2) * (1.f / 768.f));
        float th = (float)tok * fr;
        s += (d & 1) ? __cosf(th) : __sinf(th);
        h[(size_t)row * 768 + d] = s;
    }
}

// ---------------------------------------------------------------------------
// Timestep interleaved sincos embedding.  <<<B, 768>>>
// ---------------------------------------------------------------------------
__global__ void time_base_kernel(const int* __restrict__ t, float* __restrict__ te0) {
    const int b = blockIdx.x;
    const int d = threadIdx.x;
    const int i2 = d >> 1;
    float fr = __powf(10000.f, -(2.f * (float)i2) * (1.f / 768.f));
    float th = (float)t[b] * fr;
    te0[b * 768 + d] = (d & 1) ? __cosf(th) : __sinf(th);
}

// ---------------------------------------------------------------------------
// Tiny GEMM (B=8 rows) for time/modulation MLPs. act: 0=none, 1=SiLU.
// ---------------------------------------------------------------------------
__global__ void small_gemm(const float* __restrict__ A, const float* __restrict__ W,
                           const float* __restrict__ bias, float* __restrict__ C,
                           int Brows, int K, int N, int act) {
    int idx = blockIdx.x * blockDim.x + threadIdx.x;
    if (idx >= Brows * N) return;
    int b = idx / N, n = idx - b * N;
    const float* a = A + (size_t)b * K;
    float s = bias[n];
    for (int k = 0; k < K; ++k) s += a[k] * W[(size_t)k * N + n];
    if (act == 1) s = s / (1.f + __expf(-s));
    C[idx] = s;
}

// ---------------------------------------------------------------------------
// (optionally modulated) LayerNorm: out = mg * (norm(h)*g + b) + mb
// ---------------------------------------------------------------------------
__global__ __launch_bounds__(256) void ln_mod_kernel(const float* __restrict__ hin,
                                                     const float* __restrict__ g,
                                                     const float* __restrict__ bta,
                                                     const float* __restrict__ mg,
                                                     const float* __restrict__ mb,
                                                     float* __restrict__ outp,
                                                     int mstride) {
    __shared__ float red[256], red2[256];
    const int row = blockIdx.x;
    const int b   = row >> 10;
    const int tid = threadIdx.x;
    const float* hr = hin + (size_t)row * 768;
    float s = 0.f, s2 = 0.f;
    for (int d = tid; d < 768; d += 256) { float v = hr[d]; s += v; s2 += v * v; }
    red[tid] = s; red2[tid] = s2;
    __syncthreads();
    for (int st = 128; st > 0; st >>= 1) {
        if (tid < st) { red[tid] += red[tid + st]; red2[tid] += red2[tid + st]; }
        __syncthreads();
    }
    const float mean = red[0] * (1.f / 768.f);
    const float var  = red2[0] * (1.f / 768.f) - mean * mean;
    const float inv  = rsqrtf(var + 1e-5f);
    float* orow = outp + (size_t)row * 768;
    for (int d = tid; d < 768; d += 256) {
        float v = (hr[d] - mean) * inv * g[d] + bta[d];
        if (mg) v = mg[(size_t)b * mstride + d] * v + mb[(size_t)b * mstride + d];
        orow[d] = v;
    }
}

// ---------------------------------------------------------------------------
// Gated residual: h += gate[b,:] * o
// ---------------------------------------------------------------------------
__global__ void resgate_kernel(float* __restrict__ h, const float* __restrict__ o,
                               const float* __restrict__ gate, int gstride) {
    size_t idx = (size_t)blockIdx.x * blockDim.x + threadIdx.x;
    int d = (int)(idx % 768);
    int b = (int)((idx / 768) >> 10);
    h[idx] += gate[(size_t)b * gstride + d] * o[idx];
}

// ---------------------------------------------------------------------------
// Split fused qkv (f32, [B*TOK, 3*H]) into bf16 q/k/v buffers [B, NH, TOK, HD]
// ---------------------------------------------------------------------------
__global__ void split_qkv_kernel(const float* __restrict__ qkv, bf16* __restrict__ q,
                                 bf16* __restrict__ k, bf16* __restrict__ v) {
    size_t idx = (size_t)blockIdx.x * blockDim.x + threadIdx.x;
    size_t row = idx / 2304;
    int col = (int)(idx % 2304);
    int b = (int)(row >> 10), tok = (int)(row & 1023);
    int which = col / 768, hc = col % 768, head = hc >> 6, hd = hc & 63;
    bf16 val = (bf16)qkv[idx];
    size_t o = (((size_t)b * 12 + head) * 1024 + tok) * 64 + hd;
    if (which == 0) q[o] = val; else if (which == 1) k[o] = val; else v[o] = val;
}

// ---------------------------------------------------------------------------
// Unpatchify [B*TOK, 192] -> [B, C, 256, 256]
// ---------------------------------------------------------------------------
__global__ void unpatchify_kernel(const float* __restrict__ proj, float* __restrict__ out) {
    size_t idx = (size_t)blockIdx.x * blockDim.x + threadIdx.x;
    int xq = (int)(idx & 255);
    size_t r = idx >> 8;
    int y = (int)(r & 255); r >>= 8;
    int c = (int)(r % 3);
    int b = (int)(r / 3);
    int gh = y >> 3, p = y & 7, gw = xq >> 3, qq = xq & 7;
    int tok = gh * 32 + gw;
    out[idx] = proj[((size_t)(b * 1024 + tok)) * 192 + p * 24 + qq * 3 + c];
}

// ---------------------------------------------------------------------------
// bf16 WMMA GEMM with async double-buffered LDS pipeline:
//   C[M,N] = act(A[M,K] @ W[K,N] + bias)
// 128x128x32 tiles, 8 waves (4x2), each wave 2x4 of 16x16 accumulators.
// Tiles staged fp32 via global_load_async_to_lds_b128 (ASYNCcnt), converted
// to bf16 during fragment assembly, computed with v_wmma_f32_16x16x32_bf16.
// ACT: 0 = none, 1 = exact GELU, 2 = SiLU.
// Requires: M % 128 == 0, K % 32 == 0, N >= 128 (tail columns clamped+masked).
// ---------------------------------------------------------------------------
template <int ACT>
__global__ __launch_bounds__(256) void gemm_wmma(const float* __restrict__ A,
                                                 const float* __restrict__ W,
                                                 const float* __restrict__ bias,
                                                 float* __restrict__ C,
                                                 int M, int N, int K) {
    constexpr int BM = 128, BN = 128, BK = 32;
    constexpr int AKP = 36;    // fp32 row stride, 144B (16B multiple for B128)
    constexpr int BNP = 132;   // fp32 row stride, 528B (16B multiple for B128)
    __shared__ float Asb[2][BM * AKP];
    __shared__ float Bsb[2][BK * BNP];
    const int bm = blockIdx.y, bn = blockIdx.x;
    const int tid  = threadIdx.x;
    const int wave = tid >> 5, lane = tid & 31;
    const int hl   = lane >> 4;
    const int l16  = lane & 15;
    const int wrow = (wave >> 1) * 32;  // 0,32,64,96
    const int wcol = (wave & 1) * 64;   // 0,64

    // issue one K-stage: 8 async B128 copies per thread (4 A-chunks, 4 B-chunks)
    auto issue_stage = [&](int k0, float* Adst, float* Bdst) {
        #pragma unroll
        for (int i = 0; i < 4; ++i) {               // A tile: 1024 x 16B chunks
            int c4 = i * 256 + tid;
            int r = c4 >> 3, cc = (c4 & 7) * 4;     // 8 chunks per 32-float row
            uint32_t loff = lds_off_of(&Adst[r * AKP + cc]);
            async_copy_b128(loff, &A[(size_t)(bm * BM + r) * K + k0 + cc]);
        }
        #pragma unroll
        for (int i = 0; i < 4; ++i) {               // B tile: 1024 x 16B chunks
            int c4 = i * 256 + tid;
            int r = c4 >> 5, cc = (c4 & 31) * 4;    // 32 chunks per 128-float row
            int gc = bn * BN + cc;
            if (gc > N - 4) gc = N - 4;             // tail clamp (stores masked)
            uint32_t loff = lds_off_of(&Bdst[r * BNP + cc]);
            async_copy_b128(loff, &W[(size_t)(k0 + r) * N + gc]);
        }
    };

    v8f acc[2][4];
    #pragma unroll
    for (int mt = 0; mt < 2; ++mt)
        #pragma unroll
        for (int nt = 0; nt < 4; ++nt) acc[mt][nt] = zero_v8f();

    const int nsteps = K / BK;
    issue_stage(0, Asb[0], Bsb[0]);

    for (int s = 0; s < nsteps; ++s) {
        const int cur = s & 1;
        if (s + 1 < nsteps) {
            issue_stage((s + 1) * BK, Asb[cur ^ 1], Bsb[cur ^ 1]);
            // async ops retire in order per wave: <=8 outstanding means the
            // current stage's 8 copies have landed in LDS.
            asm volatile("s_wait_asynccnt 0x8" ::: "memory");
        } else {
            asm volatile("s_wait_asynccnt 0x0" ::: "memory");
        }
        __syncthreads();

        const float* Ab = Asb[cur];
        const float* Bb = Bsb[cur];
        #pragma unroll
        for (int mt = 0; mt < 2; ++mt) {
            // A fragment (16x32), fp32 LDS -> bf16, half-wave holds contiguous 16 K values
            v16bf af;
            const float* ap = &Ab[(wrow + mt * 16 + l16) * AKP];
            #pragma unroll
            for (int i = 0; i < 8; ++i) {
                af[i]     = (bf16)ap[hl * 16 + i];
                af[8 + i] = (bf16)ap[hl * 16 + 8 + i];
            }
            #pragma unroll
            for (int nt = 0; nt < 4; ++nt) {
                // B fragment (32x16): half-wave holds contiguous 16 K values
                v16bf bfv;
                const float* bp = &Bb[wcol + nt * 16 + l16];
                #pragma unroll
                for (int j = 0; j < 16; ++j) bfv[j] = (bf16)bp[(hl * 16 + j) * BNP];
                acc[mt][nt] = __builtin_amdgcn_wmma_f32_16x16x32_bf16(
                    false, af, false, bfv, (short)0, acc[mt][nt], false, false);
            }
        }
        __syncthreads();   // all waves done with `cur` before it is re-issued
    }

    // epilogue: bias + activation
    #pragma unroll
    for (int mt = 0; mt < 2; ++mt) {
        #pragma unroll
        for (int nt = 0; nt < 4; ++nt) {
            int col = bn * BN + wcol + nt * 16 + l16;
            float bv = (col < N) ? bias[col] : 0.f;
            #pragma unroll
            for (int r = 0; r < 8; ++r) {
                int row = bm * BM + wrow + mt * 16 + r + 8 * hl;
                if (row < M && col < N) {
                    float v = acc[mt][nt][r] + bv;
                    if (ACT == 1)      v = 0.5f * v * (1.f + erff(v * 0.70710678118f));
                    else if (ACT == 2) v = v / (1.f + __expf(-v));
                    C[(size_t)row * N + col] = v;
                }
            }
        }
    }
}

// ---------------------------------------------------------------------------
// Fused flash attention.  grid = (TOK/128, NH, B), 256 threads (8 waves).
// Per wave: 16-row query tile, 32 keys/iter: 4 WMMAs S=QK^T, online softmax,
// P relayout through per-wave LDS, 4 WMMAs O += P V.
// ---------------------------------------------------------------------------
__global__ __launch_bounds__(256) void attn_kernel(const bf16* __restrict__ Q,
                                                   const bf16* __restrict__ Km,
                                                   const bf16* __restrict__ V,
                                                   float* __restrict__ O) {
    const int qtile = blockIdx.x;
    const int head  = blockIdx.y;
    const int b     = blockIdx.z;
    const int wave  = threadIdx.x >> 5;
    const int lane  = threadIdx.x & 31;
    const int hl    = lane >> 4;
    const int l16   = lane & 15;
    const int q0    = qtile * 128 + wave * 16;

    const size_t hb = ((size_t)b * 12 + head) * 1024;
    const bf16* qp = Q  + (hb + q0) * 64;
    const bf16* kp = Km + hb * 64;
    const bf16* vp = V  + hb * 64;

    __shared__ bf16 p_lds[8][16 * 32];

    v16bf qa0, qa1;
    {
        const bf16* qrow = qp + (size_t)l16 * 64;
        #pragma unroll
        for (int i = 0; i < 8; ++i) {
            qa0[i]     = qrow[hl * 16 + i];
            qa0[8 + i] = qrow[hl * 16 + 8 + i];
            qa1[i]     = qrow[32 + hl * 16 + i];
            qa1[8 + i] = qrow[40 + hl * 16 + i];
        }
    }

    float m_i[8], l_i[8];
    v8f o_acc[4];
    #pragma unroll
    for (int r = 0; r < 8; ++r) { m_i[r] = -1e30f; l_i[r] = 0.f; }
    #pragma unroll
    for (int nt = 0; nt < 4; ++nt) o_acc[nt] = zero_v8f();

    const float scale = 0.03608439182f;   // 1/sqrt(768)

    for (int s0 = 0; s0 < 1024; s0 += 32) {
        v8f sc0 = zero_v8f(), sc1 = zero_v8f();
        {   // keys s0 .. s0+15
            const bf16* krow = kp + (size_t)(s0 + l16) * 64;
            v16bf kb;
            #pragma unroll
            for (int j = 0; j < 16; ++j) kb[j] = krow[hl * 16 + j];
            sc0 = __builtin_amdgcn_wmma_f32_16x16x32_bf16(false, qa0, false, kb, (short)0, sc0, false, false);
            #pragma unroll
            for (int j = 0; j < 16; ++j) kb[j] = krow[32 + hl * 16 + j];
            sc0 = __builtin_amdgcn_wmma_f32_16x16x32_bf16(false, qa1, false, kb, (short)0, sc0, false, false);
        }
        {   // keys s0+16 .. s0+31
            const bf16* krow = kp + (size_t)(s0 + 16 + l16) * 64;
            v16bf kb;
            #pragma unroll
            for (int j = 0; j < 16; ++j) kb[j] = krow[hl * 16 + j];
            sc1 = __builtin_amdgcn_wmma_f32_16x16x32_bf16(false, qa0, false, kb, (short)0, sc1, false, false);
            #pragma unroll
            for (int j = 0; j < 16; ++j) kb[j] = krow[32 + hl * 16 + j];
            sc1 = __builtin_amdgcn_wmma_f32_16x16x32_bf16(false, qa1, false, kb, (short)0, sc1, false, false);
        }
        // online softmax per row (row r+8*hl lives in vgpr r across a 16-lane half)
        #pragma unroll
        for (int r = 0; r < 8; ++r) {
            float v0 = sc0[r] * scale;
            float v1 = sc1[r] * scale;
            float mx = fmaxf(v0, v1);
            #pragma unroll
            for (int off = 8; off >= 1; off >>= 1) mx = fmaxf(mx, __shfl_xor(mx, off, 32));
            float mnew = fmaxf(m_i[r], mx);
            float corr = __expf(m_i[r] - mnew);
            float p0 = __expf(v0 - mnew);
            float p1 = __expf(v1 - mnew);
            float rs = p0 + p1;
            #pragma unroll
            for (int off = 8; off >= 1; off >>= 1) rs += __shfl_xor(rs, off, 32);
            l_i[r] = l_i[r] * corr + rs;
            m_i[r] = mnew;
            #pragma unroll
            for (int nt = 0; nt < 4; ++nt) o_acc[nt][r] *= corr;
            int row = r + 8 * hl;
            p_lds[wave][row * 32 + l16]      = (bf16)p0;
            p_lds[wave][row * 32 + 16 + l16] = (bf16)p1;
        }
        __syncthreads();
        v16bf pa;
        {
            const bf16* prow = &p_lds[wave][l16 * 32];
            #pragma unroll
            for (int i = 0; i < 8; ++i) {
                pa[i]     = prow[hl * 16 + i];
                pa[8 + i] = prow[hl * 16 + 8 + i];
            }
        }
        #pragma unroll
        for (int nt = 0; nt < 4; ++nt) {
            v16bf vbf;
            #pragma unroll
            for (int j = 0; j < 16; ++j)
                vbf[j] = vp[(size_t)(s0 + hl * 16 + j) * 64 + nt * 16 + l16];
            o_acc[nt] = __builtin_amdgcn_wmma_f32_16x16x32_bf16(
                false, pa, false, vbf, (short)0, o_acc[nt], false, false);
        }
        __syncthreads();
    }

    #pragma unroll
    for (int r = 0; r < 8; ++r) {
        int row = r + 8 * hl;
        float inv = 1.f / l_i[r];
        size_t base = ((size_t)b * 1024 + q0 + row) * 768 + head * 64;
        #pragma unroll
        for (int nt = 0; nt < 4; ++nt)
            O[base + nt * 16 + l16] = o_acc[nt][r] * inv;
    }
}

// ---------------------------------------------------------------------------
// Host orchestration
// ---------------------------------------------------------------------------
extern "C" void kernel_launch(void* const* d_in, const int* in_sizes, int n_in,
                              void* d_out, int out_size, void* d_ws, size_t ws_size,
                              hipStream_t stream) {
    (void)in_sizes; (void)n_in; (void)out_size; (void)ws_size;
    const float* x      = (const float*)d_in[0];
    const int*   t      = (const int*)d_in[1];
    const float* conv_w = (const float*)d_in[2];
    const float* conv_b = (const float*)d_in[3];
    const float* te_w1  = (const float*)d_in[4];
    const float* te_b1  = (const float*)d_in[5];
    const float* te_w2  = (const float*)d_in[6];
    const float* te_b2  = (const float*)d_in[7];
    const float* ln1_g  = (const float*)d_in[8];
    const float* ln1_b  = (const float*)d_in[9];
    const float* ln2_g  = (const float*)d_in[10];
    const float* ln2_b  = (const float*)d_in[11];
    const float* qkv_w  = (const float*)d_in[12];
    const float* qkv_b  = (const float*)d_in[13];
    const float* mha_w1 = (const float*)d_in[14];
    const float* mha_b1 = (const float*)d_in[15];
    const float* mha_w2 = (const float*)d_in[16];
    const float* mha_b2 = (const float*)d_in[17];
    const float* ss_w1  = (const float*)d_in[18];
    const float* ss_b1  = (const float*)d_in[19];
    const float* ss_w2  = (const float*)d_in[20];
    const float* ss_b2  = (const float*)d_in[21];
    const float* ffn_w1 = (const float*)d_in[22];
    const float* ffn_b1 = (const float*)d_in[23];
    const float* ffn_w2 = (const float*)d_in[24];
    const float* ffn_b2 = (const float*)d_in[25];
    const float* lnf_g  = (const float*)d_in[26];
    const float* lnf_b  = (const float*)d_in[27];
    const float* out_w  = (const float*)d_in[28];
    const float* out_b  = (const float*)d_in[29];
    float* out = (float*)d_out;

    char* ws = (char*)d_ws;
    size_t off = 0;
    auto alloc = [&](size_t bytes) -> void* {
        void* p = ws + off;
        off += (bytes + 255) & ~(size_t)255;
        return p;
    };
    const size_t ROWS = 8192;                       // B * TOK
    float* h    = (float*)alloc(ROWS * 768 * 4);
    float* z    = (float*)alloc(ROWS * 768 * 4);
    float* qkvb = (float*)alloc(ROWS * 2304 * 4);
    bf16*  qb   = (bf16*) alloc(ROWS * 768 * 2);
    bf16*  kb   = (bf16*) alloc(ROWS * 768 * 2);
    bf16*  vb   = (bf16*) alloc(ROWS * 768 * 2);
    float* att  = (float*)alloc(ROWS * 768 * 4);
    float* mid  = (float*)alloc(ROWS * 3072 * 4);
    float* o2   = (float*)alloc(ROWS * 768 * 4);
    float* proj = (float*)alloc(ROWS * 192 * 4);
    float* te0  = (float*)alloc(8 * 768 * 4);
    float* teh  = (float*)alloc(8 * 768 * 4);
    float* te   = (float*)alloc(8 * 768 * 4);
    float* ssh  = (float*)alloc(8 * 4608 * 4);
    float* temb = (float*)alloc(8 * 4608 * 4);

    patchify_kernel<<<8192, 256, 0, stream>>>(x, conv_w, conv_b, h);
    time_base_kernel<<<8, 768, 0, stream>>>(t, te0);
    small_gemm<<<(8 * 768 + 255) / 256, 256, 0, stream>>>(te0, te_w1, te_b1, teh, 8, 768, 768, 1);
    small_gemm<<<(8 * 768 + 255) / 256, 256, 0, stream>>>(teh, te_w2, te_b2, te, 8, 768, 768, 0);

    dim3 gQKV(18, 64), g4H(24, 64), gH(6, 64), gOut(2, 64);
    for (int i = 0; i < 12; ++i) {
        small_gemm<<<(8 * 4608 + 255) / 256, 256, 0, stream>>>(
            te, ss_w1 + (size_t)i * 768 * 4608, ss_b1 + (size_t)i * 4608, ssh, 8, 768, 4608, 1);
        small_gemm<<<(8 * 4608 + 255) / 256, 256, 0, stream>>>(
            ssh, ss_w2 + (size_t)i * 4608 * 4608, ss_b2 + (size_t)i * 4608, temb, 8, 4608, 4608, 0);

        // attention branch
        ln_mod_kernel<<<8192, 256, 0, stream>>>(h, ln1_g + (size_t)i * 768, ln1_b + (size_t)i * 768,
                                                temb + 0, temb + 768, z, 4608);
        gemm_wmma<0><<<gQKV, 256, 0, stream>>>(z, qkv_w + (size_t)i * 768 * 2304,
                                               qkv_b + (size_t)i * 2304, qkvb, 8192, 2304, 768);
        split_qkv_kernel<<<(8192 * 2304) / 256, 256, 0, stream>>>(qkvb, qb, kb, vb);
        attn_kernel<<<dim3(8, 12, 8), 256, 0, stream>>>(qb, kb, vb, att);
        gemm_wmma<1><<<g4H, 256, 0, stream>>>(att, mha_w1 + (size_t)i * 768 * 3072,
                                              mha_b1 + (size_t)i * 3072, mid, 8192, 3072, 768);
        gemm_wmma<0><<<gH, 256, 0, stream>>>(mid, mha_w2 + (size_t)i * 3072 * 768,
                                             mha_b2 + (size_t)i * 768, o2, 8192, 768, 3072);
        resgate_kernel<<<(8192 * 768) / 256, 256, 0, stream>>>(h, o2, temb + 1536, 4608);

        // FFN branch
        ln_mod_kernel<<<8192, 256, 0, stream>>>(h, ln2_g + (size_t)i * 768, ln2_b + (size_t)i * 768,
                                                temb + 2304, temb + 3072, z, 4608);
        gemm_wmma<1><<<g4H, 256, 0, stream>>>(z, ffn_w1 + (size_t)i * 768 * 3072,
                                              ffn_b1 + (size_t)i * 3072, mid, 8192, 3072, 768);
        gemm_wmma<0><<<gH, 256, 0, stream>>>(mid, ffn_w2 + (size_t)i * 3072 * 768,
                                             ffn_b2 + (size_t)i * 768, o2, 8192, 768, 3072);
        resgate_kernel<<<(8192 * 768) / 256, 256, 0, stream>>>(h, o2, temb + 3840, 4608);
    }

    ln_mod_kernel<<<8192, 256, 0, stream>>>(h, lnf_g, lnf_b, nullptr, nullptr, z, 0);
    gemm_wmma<0><<<gOut, 256, 0, stream>>>(z, out_w, out_b, proj, 8192, 192, 768);
    unpatchify_kernel<<<(8 * 3 * 256 * 256) / 256, 256, 0, stream>>>(proj, out);
}
